// GroupedMultiheadSelfAttention_77644418777372
// MI455X (gfx1250) — matrix operand order repaired and checked by reading, not compiled, not run on hardware
//
#include <hip/hip_runtime.h>
#include <hip/hip_bf16.h>
#include <cmath>

typedef __attribute__((ext_vector_type(16))) __bf16 v16bf;
typedef __attribute__((ext_vector_type(8)))  float  v8f;
typedef __attribute__((ext_vector_type(4)))  unsigned int v4u;
typedef __attribute__((ext_vector_type(4)))  int v4i;

#define WMMA_BF16(a, b, c) \
    __builtin_amdgcn_wmma_f32_16x16x32_bf16(false, (a), false, (b), (short)0, (c), false, false)

static constexpr int Sq = 1024;        // sequence length
static constexpr int Bn = 8;           // batch
static constexpr int Mrows = Bn * Sq;  // 8192

// ---------------------------------------------------------------------------
// Async global->LDS copy (CDNA5 GLOBAL_LOAD_ASYNC_TO_LDS_B128, ASYNCcnt),
// with a synchronous VGPR fallback if the builtin is absent.
// Builtin signature (from compiler diagnostic): (v4i AS1* src, v4i AS3* dst,
// imm offset, imm cpol).  Address-space pointers are formed via integer casts:
// generic-LDS low 32 bits == LDS byte address (ISA 10.2), generic-global ==
// AS1 address bit-identically.
// ---------------------------------------------------------------------------
#if __has_builtin(__builtin_amdgcn_global_load_async_to_lds_b128)
#define HAS_ASYNC_LDS 1
typedef __attribute__((address_space(1))) v4i* g_v4i_p;
typedef __attribute__((address_space(3))) v4i* l_v4i_p;
#else
#define HAS_ASYNC_LDS 0
#endif

__device__ __forceinline__ void cp16(__bf16* dst_lds, const __bf16* src_g)
{
#if HAS_ASYNC_LDS
    g_v4i_p g = (g_v4i_p)(unsigned long long)src_g;
    l_v4i_p l = (l_v4i_p)(unsigned int)(unsigned long long)dst_lds;
    __builtin_amdgcn_global_load_async_to_lds_b128(g, l, 0, 0);
#else
    *(v4u*)dst_lds = *(const v4u*)src_g;
#endif
}

template <int N>
__device__ __forceinline__ void wait_async()
{
#if HAS_ASYNC_LDS
#if __has_builtin(__builtin_amdgcn_s_wait_asynccnt)
    __builtin_amdgcn_s_wait_asynccnt((short)N);
#else
    asm volatile("s_wait_asynccnt %0" ::"i"(N) : "memory");
#endif
#endif
}

// ---------------------------------------------------------------------------
// Weight pre-transpose: WT[N x K](bf16) = W[K x N](f32)^T.  K is a power of 2.
// Coalesced writes; done once per weight matrix (all weights ~8 MB total).
// ---------------------------------------------------------------------------
__global__ void prep_wT(const float* __restrict__ W, __bf16* __restrict__ WT,
                        int kmask, int kshift, int N)
{
    const int idx = blockIdx.x * 256 + threadIdx.x;   // idx = n*K + kk
    const int kk = idx & kmask;
    const int n  = idx >> kshift;
    WT[idx] = (__bf16)W[(size_t)kk * N + n];
}

// ---------------------------------------------------------------------------
// Projection GEMM: Y[M x N](bf16) = cvt_bf16(X[M x K]) @ WT^T + bias
// WT is the pre-transposed bf16 weight (N x K) -> contiguous B-frag loads.
// One wave per 16x16 tile. transposed=1 stores Y as (B, N, S).
// ---------------------------------------------------------------------------
__global__ __launch_bounds__(32)
void proj_gemm(const float* __restrict__ X, const __bf16* __restrict__ WT,
               const float* __restrict__ bias, __bf16* __restrict__ Y,
               int K, int N, int transposed)
{
    const int lane = threadIdx.x;
    const int r16  = lane & 15;
    const int hs   = lane >> 4;
    const int row  = blockIdx.y * 16 + r16;
    const int col  = blockIdx.x * 16 + r16;

    v8f acc = {0.f, 0.f, 0.f, 0.f, 0.f, 0.f, 0.f, 0.f};

    for (int k0 = 0; k0 < K; k0 += 32) {
        v16bf a, b;
        const float* ap = X + (size_t)row * K + k0 + hs * 8;
#pragma unroll
        for (int j = 0; j < 8; ++j) {
            a[j]     = (__bf16)ap[j];
            a[8 + j] = (__bf16)ap[16 + j];
        }
        const __bf16* wp = WT + (size_t)col * K + k0 + hs * 8;
#pragma unroll
        for (int j = 0; j < 8; ++j) { b[j] = wp[j]; b[8 + j] = wp[16 + j]; }
        acc = WMMA_BF16(a, b, acc);
    }

    const float bc = bias[col];
    const int rr0 = blockIdx.y * 16 + hs * 8;
    if (!transposed) {
#pragma unroll
        for (int r = 0; r < 8; ++r)
            Y[(size_t)(rr0 + r) * N + col] = (__bf16)(acc[r] + bc);
    } else {
        const int batch = rr0 >> 10;
        const int srow  = rr0 & 1023;
        __bf16* base = Y + ((size_t)batch * N + col) * Sq + srow;
#pragma unroll
        for (int r = 0; r < 8; ++r) base[r] = (__bf16)(acc[r] + bc);
    }
}

// ---------------------------------------------------------------------------
// Unnormalized flash attention, one (a,b) pair:
//   E = exp(scale * Q K^T); EV = E @ V; lsum = rowsum(E)
// Block = 4 waves (128 thr), 64 queries/block sharing K/V chunks staged in LDS
// via async global->LDS DMA, double-buffered on ASYNCcnt.
// EV overwrites Q in place (Q tile register-resident before any store).
// ---------------------------------------------------------------------------
template <int INNER>
__global__ __launch_bounds__(128)
void attn_kernel(__bf16* __restrict__ QEV, const __bf16* __restrict__ Km,
                 const __bf16* __restrict__ VT, float* __restrict__ lsum,
                 float scale)
{
    constexpr int NF  = INNER / 32;      // A-frags along inner dim
    constexpr int NV  = INNER / 16;      // output column tiles
    constexpr int KP  = INNER + 8;       // padded K-row stride (bank spread)
    constexpr int VP  = 40;              // padded V-row stride (32 + 8)
    constexpr int EP  = 40;              // padded E-row stride
    constexpr int TPR = INNER / 8;       // 16B transfers per K row
    constexpr int TPRL = (INNER == 256) ? 5 : 4;
    constexpr int STAGE_OPS = INNER / 16;  // async instrs per wave per stage

    __shared__ __bf16 Kls[2][32 * KP];   // K chunk: 32 keys x INNER
    __shared__ __bf16 Vls[2][INNER * VP];// V chunk: INNER feats x 32 keys
    __shared__ __bf16 Els[4][16 * EP];   // per-wave E tile staging

    const int tid   = threadIdx.x;
    const int lane  = tid & 31;
    const int wave  = tid >> 5;
    const int r16   = lane & 15;
    const int hs    = lane >> 4;
    const int batch = blockIdx.y;
    const int q0    = blockIdx.x * 64 + wave * 16;

    // ---- load the wave's Q tile into registers (A-frags) ----
    v16bf qf[NF];
    const __bf16* qrow = QEV + (size_t)(batch * Sq + q0 + r16) * INNER;
#pragma unroll
    for (int f = 0; f < NF; ++f) {
        const __bf16* p = qrow + f * 32 + hs * 8;
#pragma unroll
        for (int j = 0; j < 8; ++j) {
            qf[f][j]     = p[j];
            qf[f][8 + j] = p[16 + j];
        }
    }

    v8f av[NV];
#pragma unroll
    for (int n = 0; n < NV; ++n) av[n] = (v8f){0.f,0.f,0.f,0.f,0.f,0.f,0.f,0.f};
    float sAcc[8];
#pragma unroll
    for (int r = 0; r < 8; ++r) sAcc[r] = 0.f;

    // ---- K/V chunk staging (all 128 threads cooperate) ----
    auto stage = [&](int buf, int kc) {
        const __bf16* ksrc = Km + (size_t)(batch * Sq + kc) * INNER;
        __bf16* kdst = &Kls[buf][0];
#pragma unroll
        for (int i = 0; i < INNER / 32; ++i) {
            const int idx = i * 128 + tid;            // 16B-transfer index
            const int krow = idx >> TPRL;
            const int part = idx & (TPR - 1);
            cp16(kdst + krow * KP + part * 8, ksrc + (size_t)idx * 8);
        }
        const __bf16* vsrc = VT + (size_t)batch * INNER * Sq + kc;
        __bf16* vdst = &Vls[buf][0];
#pragma unroll
        for (int i = 0; i < INNER / 32; ++i) {
            const int idx  = i * 128 + tid;
            const int feat = idx >> 2;
            const int part = idx & 3;
            cp16(vdst + feat * VP + part * 8, vsrc + (size_t)feat * Sq + part * 8);
        }
    };

    stage(0, 0);
    for (int kci = 0; kci < Sq / 32; ++kci) {
        const int kc  = kci * 32;
        const int cur = kci & 1;
        const bool pre = (kc + 32 < Sq);
        if (pre) stage(cur ^ 1, kc + 32);            // prefetch next chunk
        if (pre) wait_async<STAGE_OPS>();            // current chunk done
        else     wait_async<0>();
        __syncthreads();

        // ---- logits for 32 keys (two 16x16 tiles), exp, stage E ----
#pragma unroll
        for (int kt = 0; kt < 2; ++kt) {
            v8f lg = {0.f,0.f,0.f,0.f,0.f,0.f,0.f,0.f};
            const __bf16* krow = &Kls[cur][(kt * 16 + r16) * KP + hs * 8];
#pragma unroll
            for (int f = 0; f < NF; ++f) {
                v16bf kb;
                const __bf16* p = krow + f * 32;
#pragma unroll
                for (int j = 0; j < 8; ++j) { kb[j] = p[j]; kb[8 + j] = p[16 + j]; }
                lg = WMMA_BF16(qf[f], kb, lg);
            }
#pragma unroll
            for (int r = 0; r < 8; ++r) {
                const float e = __expf(lg[r] * scale);
                sAcc[r] += e;
                Els[wave][(hs * 8 + r) * EP + kt * 16 + r16] = (__bf16)e;
            }
        }
        // E tile C-layout -> A-layout via per-wave LDS (wave-internal, in order)
        v16bf ef;
        {
            const __bf16* er = &Els[wave][r16 * EP + hs * 8];
#pragma unroll
            for (int j = 0; j < 8; ++j) {
                ef[j]     = er[j];
                ef[8 + j] = er[16 + j];
            }
        }
        // ---- EV += E @ V ----
#pragma unroll
        for (int n = 0; n < NV; ++n) {
            v16bf vb;
            const __bf16* vp = &Vls[cur][(n * 16 + r16) * VP + hs * 8];
#pragma unroll
            for (int j = 0; j < 8; ++j) { vb[j] = vp[j]; vb[8 + j] = vp[16 + j]; }
            av[n] = WMMA_BF16(ef, vb, av[n]);
        }
        __syncthreads();                              // chunk reads done
    }

    // ---- row-sums: reduce across the 16 lanes of each half-wave ----
#pragma unroll
    for (int r = 0; r < 8; ++r) {
        float v = sAcc[r];
        v += __shfl_xor(v, 1, 32);
        v += __shfl_xor(v, 2, 32);
        v += __shfl_xor(v, 4, 32);
        v += __shfl_xor(v, 8, 32);
        sAcc[r] = v;
    }
    if (r16 == 0) {
#pragma unroll
        for (int r = 0; r < 8; ++r)
            lsum[batch * Sq + q0 + hs * 8 + r] = sAcc[r];
    }

    // ---- store EV over the Q buffer ----
    __bf16* evrow = QEV + (size_t)(batch * Sq + q0 + hs * 8) * INNER;
#pragma unroll
    for (int n = 0; n < NV; ++n) {
        const int col = n * 16 + r16;
#pragma unroll
        for (int r = 0; r < 8; ++r)
            evrow[(size_t)r * INNER + col] = (__bf16)av[n][r];
    }
}

// ---------------------------------------------------------------------------
// invL[a][row] = 1 / sum_b lsum[a*4+b][row]
// ---------------------------------------------------------------------------
__global__ void invl_kernel(const float* __restrict__ lsum, float* __restrict__ invL)
{
    const int row = blockIdx.x * blockDim.x + threadIdx.x;
#pragma unroll
    for (int a = 0; a < 4; ++a) {
        float L = 0.f;
#pragma unroll
        for (int b = 0; b < 4; ++b) L += lsum[(size_t)(a * 4 + b) * Mrows + row];
        invL[(size_t)a * Mrows + row] = 1.f / L;
    }
}

// ---------------------------------------------------------------------------
// out_a[row][col] = invL_a[row] * sum_b lsum_ab[row] * bo_ab[col]
// ---------------------------------------------------------------------------
__global__ void bias_init(const float* __restrict__ l0, const float* __restrict__ l1,
                          const float* __restrict__ l2, const float* __restrict__ l3,
                          const float* __restrict__ bo0, const float* __restrict__ bo1,
                          const float* __restrict__ bo2, const float* __restrict__ bo3,
                          const float* __restrict__ invLa, float* __restrict__ out, int sh)
{
    const size_t idx = (size_t)blockIdx.x * blockDim.x + threadIdx.x;
    const size_t row = idx >> sh;
    const int col = (int)(idx & (((size_t)1 << sh) - 1));
    out[idx] = invLa[row] * (l0[row] * bo0[col] + l1[row] * bo1[col] +
                             l2[row] * bo2[col] + l3[row] * bo3[col]);
}

// ---------------------------------------------------------------------------
// Accumulate GEMM: Out[M x N] += diag(rs) * (A_bf16[M x K] @ WoT^T)
// Row scaling applied in the epilogue (commutes with the matmul).
// ---------------------------------------------------------------------------
__global__ __launch_bounds__(32)
void accum_gemm(const __bf16* __restrict__ A, const float* __restrict__ rs,
                const __bf16* __restrict__ WT, float* __restrict__ Out, int K, int N)
{
    const int lane = threadIdx.x;
    const int r16  = lane & 15;
    const int hs   = lane >> 4;
    const int row  = blockIdx.y * 16 + r16;
    const int col  = blockIdx.x * 16 + r16;

    v8f acc = {0.f, 0.f, 0.f, 0.f, 0.f, 0.f, 0.f, 0.f};
    for (int k0 = 0; k0 < K; k0 += 32) {
        v16bf a, b;
        const __bf16* ap = A + (size_t)row * K + k0 + hs * 8;
#pragma unroll
        for (int j = 0; j < 8; ++j) {
            a[j]     = ap[j];
            a[8 + j] = ap[16 + j];
        }
        const __bf16* wp = WT + (size_t)col * K + k0 + hs * 8;
#pragma unroll
        for (int j = 0; j < 8; ++j) { b[j] = wp[j]; b[8 + j] = wp[16 + j]; }
        acc = WMMA_BF16(a, b, acc);
    }
    const int rr0 = blockIdx.y * 16 + hs * 8;
#pragma unroll
    for (int r = 0; r < 8; ++r)
        Out[(size_t)(rr0 + r) * N + col] += rs[rr0 + r] * acc[r];
}

// ---------------------------------------------------------------------------
// Host launch
// ---------------------------------------------------------------------------
static inline int ilog2(int x) { int l = 0; while ((1 << l) < x) ++l; return l; }

extern "C" void kernel_launch(void* const* d_in, const int* in_sizes, int n_in,
                              void* d_out, int out_size, void* d_ws, size_t ws_size,
                              hipStream_t stream)
{
    (void)in_sizes; (void)n_in; (void)out_size; (void)ws_size;
    static const int GF_[4] = {512, 256, 512, 256};
    const float* words[4] = {(const float*)d_in[0], (const float*)d_in[1],
                             (const float*)d_in[2], (const float*)d_in[3]};
    // params flattened in insertion order: per pair p=a*4+b: Wq,bq,Wk,bk,Wv,bv,Wo,bo
    auto P = [&](int p, int j) { return (const float*)d_in[4 + p * 8 + j]; };

    int innerT[16], pref[17];
    pref[0] = 0;
    for (int a = 0; a < 4; ++a)
        for (int b = 0; b < 4; ++b) {
            const int p = a * 4 + b;
            innerT[p] = (GF_[a] < GF_[b] ? GF_[a] : GF_[b]) / 2;
            pref[p + 1] = pref[p] + innerT[p];
        }

    const size_t M  = Mrows;
    const size_t qe = M * (size_t)pref[16];
    __bf16* Qb = (__bf16*)d_ws;            // Q, later overwritten by EV
    __bf16* Kb = Qb + qe;
    __bf16* Vb = Kb + qe;                  // V stored transposed (B, inner, S)
    float*  lsum = (float*)(Vb + qe);      // 16 x M
    float*  invL = lsum + 16 * M;          // 4 x M
    __bf16* Wt = (__bf16*)(invL + 4 * M);  // transposed bf16 weights pool

    // weight matrix dims per pair: {Wq: GFa x in}, {Wk: GFb x in}, {Wv: GFb x in}, {Wo: in x GFa}
    size_t wOff[16][4];
    {
        size_t cur = 0;
        for (int p = 0; p < 16; ++p) {
            const int a = p / 4, b = p % 4, in_ = innerT[p];
            const int kd[4] = {GF_[a], GF_[b], GF_[b], in_};
            const int nd[4] = {in_, in_, in_, GF_[a]};
            for (int m = 0; m < 4; ++m) { wOff[p][m] = cur; cur += (size_t)kd[m] * nd[m]; }
        }
    }

    // ---- pre-transpose all weights to bf16 (N-major) ----
    for (int p = 0; p < 16; ++p) {
        const int a = p / 4, b = p % 4, in_ = innerT[p];
        const int kd[4] = {GF_[a], GF_[b], GF_[b], in_};
        const int nd[4] = {in_, in_, in_, GF_[a]};
        const int src[4] = {0, 2, 4, 6};   // Wq, Wk, Wv, Wo
        for (int m = 0; m < 4; ++m) {
            const int K = kd[m], N = nd[m];
            prep_wT<<<(K * N) / 256, 256, 0, stream>>>(P(p, src[m]), Wt + wOff[p][m],
                                                       K - 1, ilog2(K), N);
        }
    }

    // ---- Q/K/V projections (bf16 outputs, fragment-friendly layouts) ----
    for (int p = 0; p < 16; ++p) {
        const int a = p / 4, b = p % 4, in_ = innerT[p];
        __bf16* Qp = Qb + M * (size_t)pref[p];
        __bf16* Kp = Kb + M * (size_t)pref[p];
        __bf16* Vp = Vb + M * (size_t)pref[p];
        dim3 grid(in_ / 16, (int)(M / 16));
        proj_gemm<<<grid, 32, 0, stream>>>(words[a], Wt + wOff[p][0], P(p, 1), Qp, GF_[a], in_, 0);
        proj_gemm<<<grid, 32, 0, stream>>>(words[b], Wt + wOff[p][1], P(p, 3), Kp, GF_[b], in_, 0);
        proj_gemm<<<grid, 32, 0, stream>>>(words[b], Wt + wOff[p][2], P(p, 5), Vp, GF_[b], in_, 1);
    }

    // ---- attention (EV overwrites Q in place; lsum = softmax denominators) ----
    for (int p = 0; p < 16; ++p) {
        __bf16* Qp = Qb + M * (size_t)pref[p];
        __bf16* Kp = Kb + M * (size_t)pref[p];
        __bf16* Vp = Vb + M * (size_t)pref[p];
        const float scale = 1.0f / sqrtf((float)innerT[p]);
        dim3 g(Sq / 64, Bn);
        if (innerT[p] == 256)
            attn_kernel<256><<<g, 128, 0, stream>>>(Qp, Kp, Vp, lsum + (size_t)p * M, scale);
        else
            attn_kernel<128><<<g, 128, 0, stream>>>(Qp, Kp, Vp, lsum + (size_t)p * M, scale);
    }

    // ---- combine weights ----
    invl_kernel<<<(int)(M / 256), 256, 0, stream>>>(lsum, invL);

    // ---- output: weighted bias init, then accumulate (invL ⊙ EV) @ Wo per b ----
    size_t outOff = 0;
    for (int a = 0; a < 4; ++a) {
        float* outA = (float*)d_out + outOff;
        const int GFa = GF_[a];
        const int sh = (GFa == 512) ? 9 : 8;
        bias_init<<<(int)((M * GFa) / 256), 256, 0, stream>>>(
            lsum + (size_t)(a * 4 + 0) * M, lsum + (size_t)(a * 4 + 1) * M,
            lsum + (size_t)(a * 4 + 2) * M, lsum + (size_t)(a * 4 + 3) * M,
            P(a * 4 + 0, 7), P(a * 4 + 1, 7), P(a * 4 + 2, 7), P(a * 4 + 3, 7),
            invL + (size_t)a * M, outA, sh);
        for (int b = 0; b < 4; ++b) {
            const int p = a * 4 + b;
            dim3 grid(GFa / 16, (int)(M / 16));
            accum_gemm<<<grid, 32, 0, stream>>>(Qb + M * (size_t)pref[p],
                                                invL + (size_t)a * M,
                                                Wt + wOff[p][3], outA, innerT[p], GFa);
        }
        outOff += M * (size_t)GFa;
    }
}
